// _DenseLayer_89790586290751
// MI455X (gfx1250) — compile-verified, matrix operand order checked by reading, not run on hardware
//
#include <hip/hip_runtime.h>

typedef __attribute__((ext_vector_type(16))) _Float16 v16h;
typedef __attribute__((ext_vector_type(8)))  float    v8f;

#define B_       32
#define HH       56
#define WW       56
#define HW       3136
#define NIF      64
#define GR       32
#define FEAT0    (64*32*3136)       // offset of conv feats inside d_out (after x)
#define FEATSTEP (32*32*3136)       // one 32-channel feat tensor
#define IDXSTRIDE 256
#define PXB      224                // pixels per block (3136 = 14*224)
#define HALO     57                 // one row + one col of halo each side
#define STAGE    344                // staged pixels (>= 224 + 2*57), padded
#define NPAIR    16                 // 32 channels as 16 f16-pairs

// Map logical x_cat element (b, c, hw) to an offset inside d_out.
__device__ __forceinline__ size_t xcat_off(int b, int c, int hw) {
  if (c < NIF) return ((size_t)(b * NIF + c)) * HW + hw;
  int f = (c - NIF) >> 5;
  return (size_t)FEAT0 + (size_t)f * FEATSTEP +
         ((size_t)(b * GR + ((c - NIF) & 31))) * HW + hw;
}

// -------- identity channel indices for the first two convs --------
__global__ void k_idx_identity(int* __restrict__ idx) {
  int t = blockIdx.x * blockDim.x + threadIdx.x;
  if (t < B_ * 96) {
    int b = t / 96, c = t - b * 96;
    idx[b * IDXSTRIDE + c] = c;
  }
}

// -------- pack conv weights into WMMA B fragments (f16) --------
// K-tiling: kt = cseg*9 + r ; within-tile koff (0..31) is the channel offset
// inside segment cseg. Fragment packing mirrors the ISA 16-bit operand:
// koff = (v<4?0:16) + (lane>>4)*8 + (v&3)*2 + j.
__global__ void k_pack_w(const float* __restrict__ w, unsigned* __restrict__ pB, int Cin) {
  int kt   = blockIdx.x;
  int nt   = threadIdx.x >> 5;
  int lane = threadIdx.x & 31;
  int n = lane & 15, half = lane >> 4;
  int o = nt * 16 + n;
  int cseg = kt / 9, r = kt - cseg * 9;
  unsigned res[8];
#pragma unroll
  for (int v = 0; v < 8; ++v) {
    unsigned dw = 0;
#pragma unroll
    for (int j = 0; j < 2; ++j) {
      int koff = ((v & 4) ? 16 : 0) + half * 8 + (v & 3) * 2 + j;
      int csel = cseg * 32 + koff;
      float val = w[((size_t)o * Cin + csel) * 9 + r];
      union { _Float16 f; unsigned short u; } cv;
      cv.f = (_Float16)val;
      dw |= ((unsigned)cv.u) << (16 * j);
    }
    res[v] = dw;
  }
  unsigned* dst = pB + ((size_t)(kt * 2 + nt) * 32 + lane) * 8;
#pragma unroll
  for (int v = 0; v < 8; ++v) dst[v] = res[v];
}

// -------- per-layer segment bases: segOff[b][cseg] = base of contiguous
// 32-channel run this segment gathers (gate groups are contiguous & aligned).
__global__ void k_conv_meta(const int* __restrict__ idx, int* __restrict__ segOff,
                            int NSEG) {
  int t = blockIdx.x * blockDim.x + threadIdx.x;
  if (t >= 32 * NSEG) return;
  int b = t / NSEG, s = t - b * NSEG;
  int ch0 = idx[b * IDXSTRIDE + s * 32];
  segOff[(b << 3) + s] = (int)xcat_off(b, ch0, 0);
}

// -------- implicit-GEMM conv3x3 via V_WMMA_F32_16X16X32_F16 --------
// Grid (14, B): block = 224 threads = 7 waves, 32 pixels/wave (one sample).
// Per channel-segment: stage (32ch x 344px) halo tile into LDS as f16 pairs,
// then run all 9 kernel positions from LDS (36 WMMAs per staging round).
__global__ __launch_bounds__(PXB)
void k_conv_wmma(const float* __restrict__ xcat, float* __restrict__ out,
                 const unsigned* __restrict__ pB, const int* __restrict__ segOff,
                 int NSEG) {
  __shared__ unsigned ldsA[NPAIR * STAGE];   // 22016 bytes

  int lane = threadIdx.x & 31;
  int wave = threadIdx.x >> 5;
  int b = blockIdx.y;
  int pinStart = blockIdx.x * PXB;
  int m = lane & 15, half = lane >> 4;
  int lp0 = wave * 32 + m;                   // local pixel, m-tile 0
  int lp1 = lp0 + 16;                        // m-tile 1
  int pin0 = pinStart + lp0, pin1 = pinStart + lp1;
  int h0 = pin0 / WW, w0 = pin0 - h0 * WW;
  int h1 = pin1 / WW, w1 = pin1 - h1 * WW;
  int hbase = half * 4 * STAGE;              // upper lane-half -> pairs +4
  const int* segb = segOff + (b << 3);

  v8f acc00 = {0.f,0.f,0.f,0.f,0.f,0.f,0.f,0.f};
  v8f acc01 = acc00, acc10 = acc00, acc11 = acc00;

  for (int cseg = 0; cseg < NSEG; ++cseg) {
    __syncthreads();                          // protect previous round's reads
    // ---- stage 32 channels x 344 halo pixels as packed f16 pairs ----
    const float* src = xcat + segb[cseg];
#pragma unroll
    for (int pair = 0; pair < NPAIR; ++pair) {
      const float* pc = src + (size_t)(2 * pair) * HW;
      for (int sp = threadIdx.x; sp < STAGE; sp += PXB) {
        int g = pinStart - HALO + sp;
        float v0 = 0.f, v1 = 0.f;
        if ((unsigned)g < (unsigned)HW) { v0 = pc[g]; v1 = pc[g + HW]; }
        union { _Float16 f[2]; unsigned u; } pk;
        pk.f[0] = (_Float16)v0; pk.f[1] = (_Float16)v1;
        ldsA[pair * STAGE + sp] = pk.u;
      }
    }
    __syncthreads();

    // ---- 9 kernel positions from LDS ----
#pragma unroll
    for (int r = 0; r < 9; ++r) {
      const int dh = r / 3 - 1, dw = r % 3 - 1;
      bool vv0 = ((unsigned)(h0 + dh) < (unsigned)HH) &
                 ((unsigned)(w0 + dw) < (unsigned)WW);
      bool vv1 = ((unsigned)(h1 + dh) < (unsigned)HH) &
                 ((unsigned)(w1 + dw) < (unsigned)WW);
      int shift = dh * WW + dw + HALO;
      int ad0 = hbase + lp0 + shift;
      int ad1 = hbase + lp1 + shift;

      // B fragments (L2-resident packed weights)
      int kt = cseg * 9 + r;
      const uint4* q0 = (const uint4*)(pB + ((size_t)(kt * 2 + 0) * 32 + lane) * 8);
      const uint4* q1 = (const uint4*)(pB + ((size_t)(kt * 2 + 1) * 32 + lane) * 8);
      union BU { uint4 u[2]; v16h h16; } bf0, bf1;
      bf0.u[0] = q0[0]; bf0.u[1] = q0[1];
      bf1.u[0] = q1[0]; bf1.u[1] = q1[1];

      // A fragments: 8 dword DS loads each, immediate offsets, no converts
      union AU { unsigned u[8]; v16h h16; } a0, a1;
      if (vv0) {
#pragma unroll
        for (int v = 0; v < 8; ++v) {
          const int C = (v < 4) ? v : (v + 4);   // pair column 0..3, 8..11
          a0.u[v] = ldsA[ad0 + C * STAGE];
        }
      } else {
#pragma unroll
        for (int v = 0; v < 8; ++v) a0.u[v] = 0u;
      }
      if (vv1) {
#pragma unroll
        for (int v = 0; v < 8; ++v) {
          const int C = (v < 4) ? v : (v + 4);
          a1.u[v] = ldsA[ad1 + C * STAGE];
        }
      } else {
#pragma unroll
        for (int v = 0; v < 8; ++v) a1.u[v] = 0u;
      }

      acc00 = __builtin_amdgcn_wmma_f32_16x16x32_f16(false, a0.h16, false, bf0.h16,
                                                     (short)0, acc00, false, false);
      acc01 = __builtin_amdgcn_wmma_f32_16x16x32_f16(false, a0.h16, false, bf1.h16,
                                                     (short)0, acc01, false, false);
      acc10 = __builtin_amdgcn_wmma_f32_16x16x32_f16(false, a1.h16, false, bf0.h16,
                                                     (short)0, acc10, false, false);
      acc11 = __builtin_amdgcn_wmma_f32_16x16x32_f16(false, a1.h16, false, bf1.h16,
                                                     (short)0, acc11, false, false);
    }
  }

  // D layout: lane n = lane&15; VGPR rr -> pixel row m = 8*(lane>>4) + rr
  int n = lane & 15;
  int mb = half * 8;
#pragma unroll
  for (int rr = 0; rr < 8; ++rr) {
    int pinA = pinStart + wave * 32 + mb + rr;  // m-tile 0 pixel
    int pinB = pinA + 16;                       // m-tile 1 pixel
    out[((size_t)(b * GR + n)) * HW + pinA]      = acc00[rr];
    out[((size_t)(b * GR + 16 + n)) * HW + pinA] = acc01[rr];
    out[((size_t)(b * GR + n)) * HW + pinB]      = acc10[rr];
    out[((size_t)(b * GR + 16 + n)) * HW + pinB] = acc11[rr];
  }
}

// -------- per-(b,c) spatial sum / sumsq (feeds BN stats + SE means) --------
__global__ void k_chan_sums(const float* __restrict__ xcat, float* __restrict__ sum_hw,
                            float* __restrict__ sumsq_hw, int chStart, int nCh) {
  int b = blockIdx.x / nCh;
  int c = chStart + blockIdx.x % nCh;
  const float* p = xcat + xcat_off(b, c, 0);
  __shared__ float rs[256], rq[256];
  float s = 0.f, s2 = 0.f;
  for (int i = threadIdx.x; i < HW; i += 256) { float v = p[i]; s += v; s2 += v * v; }
  rs[threadIdx.x] = s; rq[threadIdx.x] = s2;
  __syncthreads();
  for (int st = 128; st > 0; st >>= 1) {
    if (threadIdx.x < st) { rs[threadIdx.x] += rs[threadIdx.x + st];
                            rq[threadIdx.x] += rq[threadIdx.x + st]; }
    __syncthreads();
  }
  if (threadIdx.x == 0) {
    sum_hw[b * 256 + c]   = rs[0];
    sumsq_hw[b * 256 + c] = rq[0];
  }
}

// -------- BN-stats + SE gate + stable ascending arg-select, one block --------
__global__ __launch_bounds__(256)
void k_gate(const float* __restrict__ sum_hw, const float* __restrict__ sumsq_hw,
            const float* __restrict__ gamma, const float* __restrict__ beta,
            const float* __restrict__ fc1, const float* __restrict__ fc2,
            int ch, int count, int* __restrict__ idx_out) {
  __shared__ float sh_m[256], sh_a[256], sh_b[256];
  __shared__ float s_sh[32 * 256];
  __shared__ float h_sh[32 * 64];
  __shared__ float sc_sh[32 * 8];
  int tid = threadIdx.x;
  int nh = ch / 4;
  int actual = (count - 1) / 2 + 1;
  const float invBHW = 1.0f / (32.0f * 3136.0f);

  for (int c = tid; c < ch; c += 256) {
    float S = 0.f, S2 = 0.f;
    for (int b = 0; b < 32; ++b) { S += sum_hw[b * 256 + c]; S2 += sumsq_hw[b * 256 + c]; }
    float mm = S * invBHW;
    float var = S2 * invBHW - mm * mm;
    float rsv = rsqrtf(var + 1e-5f);
    sh_m[c] = mm; sh_a[c] = gamma[c] * rsv; sh_b[c] = beta[c];
  }
  __syncthreads();
  for (int t = tid; t < 32 * ch; t += 256) {
    int b = t / ch, c = t - b * ch;
    float mu = sum_hw[b * 256 + c] * (1.0f / 3136.0f);
    s_sh[b * 256 + c] = sh_a[c] * (mu - sh_m[c]) + sh_b[c];
  }
  __syncthreads();
  for (int t = tid; t < 32 * nh; t += 256) {
    int b = t / nh, r = t - b * nh;
    float d = 0.f;
    for (int c = 0; c < ch; ++c) d += s_sh[b * 256 + c] * fc1[(size_t)r * ch + c];
    h_sh[b * 64 + r] = d > 0.f ? d : 0.f;
  }
  __syncthreads();
  for (int t = tid; t < 32 * count; t += 256) {
    int b = t / count, k = t - b * count;
    float d = 0.f;
    for (int r = 0; r < nh; ++r) d += h_sh[b * 64 + r] * fc2[(size_t)k * nh + r];
    sc_sh[b * 8 + k] = 1.0f / (1.0f + __expf(-d));
  }
  __syncthreads();
  if (tid < 32) {
    int b = tid;
    bool used[8] = {false, false, false, false, false, false, false, false};
    for (int c = 0; c < 64; ++c) idx_out[b * 256 + c] = c;
    for (int a = 0; a < actual; ++a) {       // stable ascending selection
      int best = -1; float bv = 0.f;
      for (int k = 0; k < count; ++k)
        if (!used[k] && (best < 0 || sc_sh[b * 8 + k] < bv)) { bv = sc_sh[b * 8 + k]; best = k; }
      used[best] = true;
      for (int q = 0; q < 32; ++q)
        idx_out[b * 256 + 64 + a * 32 + q] = 64 + best * 32 + q;
    }
  }
}

extern "C" void kernel_launch(void* const* d_in, const int* in_sizes, int n_in,
                              void* d_out, int out_size, void* d_ws, size_t ws_size,
                              hipStream_t stream) {
  (void)in_sizes; (void)n_in; (void)out_size; (void)ws_size;
  const float* x      = (const float*)d_in[0];
  const float* init_w = (const float*)d_in[1];
  const float* lw[5]; for (int i = 0; i < 5; ++i) lw[i] = (const float*)d_in[2 + i];
  const float* gm[5]; for (int i = 0; i < 5; ++i) gm[i] = (const float*)d_in[7 + i];
  const float* bt[5]; for (int i = 0; i < 5; ++i) bt[i] = (const float*)d_in[12 + i];
  const float* f1[5]; for (int i = 0; i < 5; ++i) f1[i] = (const float*)d_in[17 + i];
  const float* f2[5]; for (int i = 0; i < 5; ++i) f2[i] = (const float*)d_in[22 + i];
  float* out = (float*)d_out;

  // workspace: idx | sum_hw | sumsq_hw | segOff | packed weights (~210 KB)
  int*      idxbuf   = (int*)d_ws;
  float*    sum_hw   = (float*)((char*)d_ws + 32768);
  float*    sumsq_hw = (float*)((char*)d_ws + 65536);
  int*      segOff   = (int*)((char*)d_ws + 98304);     // 32*8 ints
  unsigned* packedB  = (unsigned*)((char*)d_ws + 114688);

  // feats[0] = x lives at the front of d_out; all kernels read x_cat from d_out
  hipMemcpyAsync(out, x, (size_t)B_ * NIF * HW * sizeof(float),
                 hipMemcpyDeviceToDevice, stream);
  k_idx_identity<<<(B_ * 96 + 255) / 256, 256, 0, stream>>>(idxbuf);
  k_chan_sums<<<B_ * 64, 256, 0, stream>>>(out, sum_hw, sumsq_hw, 0, 64);

  const int    Cin[6] = {64, 96, 96, 128, 128, 160};
  const float* Wt[6]  = {init_w, lw[0], lw[1], lw[2], lw[3], lw[4]};
  for (int L = 0; L < 6; ++L) {
    int NSEG = Cin[L] / 32;
    int KT = NSEG * 9;
    k_pack_w<<<KT, 64, 0, stream>>>(Wt[L], packedB, Cin[L]);
    k_conv_meta<<<1, 256, 0, stream>>>(idxbuf, segOff, NSEG);
    float* outL = out + (size_t)FEAT0 + (size_t)L * FEATSTEP;
    k_conv_wmma<<<dim3(HW / PXB, B_), PXB, 0, stream>>>(out, outL, packedB,
                                                        segOff, NSEG);
    if (L <= 4)  // last conv's stats/gate are dead code in the reference
      k_chan_sums<<<B_ * 32, 256, 0, stream>>>(out, sum_hw, sumsq_hw, 64 + L * 32, 32);
    if (L >= 1 && L <= 4) {
      int i = L - 1;
      int ch = 64 + 32 * (i + 2);
      int count = i + 2;
      k_gate<<<1, 256, 0, stream>>>(sum_hw, sumsq_hw, gm[i], bt[i], f1[i], f2[i],
                                    ch, count, idxbuf);
    }
  }
}
